// ConvergedInhibition_83425444757796
// MI455X (gfx1250) — matrix operand order repaired and checked by reading, not compiled, not run on hardware
//
#include <hip/hip_runtime.h>

// ---------------------------------------------------------------------------
// ConvergedInhibition on MI455X (gfx1250, wave32):
//   y[n,:,h,w] = ifft(fft(x)/fft(k)).real  ==  circulant GEMM  Y = M * X
//   with M[c][j] = g[(c-j) mod 512], g = ifft(1/fft(k)).
// GEMM: [512 x 512] * [512 x 100352], f16 WMMA with f32 accumulation.
// HBM floor ~411 MB / 23.3 TB/s ~= 17.6 us; 52.6 GFLOP on the f16 matrix
// pipe lands at the same order -> near-roofline design.
// ---------------------------------------------------------------------------

typedef __attribute__((ext_vector_type(16))) _Float16 v16h;
typedef __attribute__((ext_vector_type(8)))  _Float16 v8h;
typedef __attribute__((ext_vector_type(8)))  float    v8f;

#define CCH   512          // channels (GEMM M and K)
#define HWPIX 3136         // 56*56
#define TWO_PI 6.283185307179586f

// LDS X tile, column-major: word[col*XS_CSTRIDE + vrow], vrow = k>>1.
// 260 % 64 == 4 -> b128 fragment reads are bank-conflict-free per half-wave.
#define XS_CSTRIDE 260     // words per pixel column (256 + 4 pad)

struct W8 { int4 a, b; };  // 32 B, for bit_cast into v16h

// ---------------------------------------------------------------------------
// Kernel 1: g = ifft(1/fft(k)) via naive 512-pt DFTs (one block, 512 thr).
// O(C^2) = 262k MACs total -- negligible.
// ---------------------------------------------------------------------------
__global__ __launch_bounds__(512) void build_inv_filter(
    const float* __restrict__ kin, float* __restrict__ g)
{
    __shared__ float kr[CCH];
    __shared__ float iGre[CCH];
    __shared__ float iGim[CCH];

    const int f = threadIdx.x;
    kr[f] = kin[f];
    __syncthreads();

    // G[f] = sum_j k[j] * exp(-2*pi*i*f*j/512)
    float re = 0.0f, im = 0.0f;
    for (int j = 0; j < CCH; ++j) {
        const int   t = (f * j) & (CCH - 1);
        const float ang = -TWO_PI * (float)t / (float)CCH;
        float s, c;
        __sincosf(ang, &s, &c);
        re = fmaf(kr[j], c, re);
        im = fmaf(kr[j], s, im);
    }
    const float d = re * re + im * im;
    iGre[f] =  re / d;          // 1/G = conj(G)/|G|^2
    iGim[f] = -im / d;
    __syncthreads();

    // g[j] = (1/512) * sum_f Re( invG[f] * exp(+2*pi*i*f*j/512) )
    float acc = 0.0f;
    for (int fi = 0; fi < CCH; ++fi) {
        const int   t = (fi * f) & (CCH - 1);
        const float ang = TWO_PI * (float)t / (float)CCH;
        float s, c;
        __sincosf(ang, &s, &c);
        acc += iGre[fi] * c - iGim[fi] * s;
    }
    g[f] = acc * (1.0f / (float)CCH);
}

// ---------------------------------------------------------------------------
// Kernel 2: materialize circulant matrix M[c][j] = (f16) g[(c-j) & 511],
// row-major [512 x 512] -> 512 KB, lives in the 192 MB L2 for the whole GEMM.
// ---------------------------------------------------------------------------
__global__ __launch_bounds__(256) void build_circulant(
    const float* __restrict__ g, _Float16* __restrict__ M)
{
    const int c = blockIdx.x;
    for (int j = threadIdx.x; j < CCH; j += 256) {
        M[c * CCH + j] = (_Float16)g[(c - j) & (CCH - 1)];
    }
}

// ---------------------------------------------------------------------------
// Kernel 3: circulant GEMM.
//  Block: 256 threads = 8 waves; owns 32 pixels x all 512 channels, so X is
//  read from HBM exactly once.
//  X tile (512 K x 32 px) staged to LDS as f16 k-pair words, column-major:
//      word[col*260 + (k>>1)] = { X[2v][col] (lo), X[2v+1][col] (hi) }
//  B fragment in this kernel's contiguous k order (lanes 0-15: K 0-15, lanes 16-31:
//  K 16-31, 2 K per VGPR) = two in-order ds_load_b128 per fragment, zero
//  register shuffles.
//  A fragments from row-major M in the same contiguous k order as B:
//      lane<16 : row c0+l,    K kb+0..7 (VGPR0-3), kb+8..15 (VGPR4-7)
//      lane>=16: row c0+l-16, K kb+16..23,          kb+24..31
//  Each wave: 4 channel tiles x 2 pixel tiles = 8 f32 accumulators,
//  16 K-steps -> 128 v_wmma per wave.
// ---------------------------------------------------------------------------
__global__ __launch_bounds__(256) void circulant_gemm(
    const float* __restrict__ X, const _Float16* __restrict__ M,
    float* __restrict__ Y)
{
    __shared__ int xs32[32 * XS_CSTRIDE];     // 33,280 B

    const int blk = blockIdx.x;               // 32 pixels per block
    const int n   = (blk * 32) / HWPIX;       // HWPIX % 32 == 0 -> single image
    const int hw0 = (blk * 32) % HWPIX;
    const size_t base = (size_t)n * CCH * HWPIX + (size_t)hw0;

    // ---- stage X tile: 512 x 32 f32 -> LDS packed-f16 words (transposed) ----
    // Task (vrow, jv): float4 from rows 2vrow, 2vrow+1 at pixel quad jv;
    // pack pairwise (v_cvt_pk_rtz_f16_f32); 4 ds_store_b32, stride 260 words.
    const int tid = threadIdx.x;
    #pragma unroll
    for (int i = 0; i < 8; ++i) {
        const int q    = tid + 256 * i;       // 2048 tasks
        const int vrow = q >> 3;              // k-pair row 0..255
        const int jv   = q & 7;               // pixel quad 0..7
        const float* p0 = X + base + (size_t)(2 * vrow)     * HWPIX + jv * 4;
        const float* p1 = X + base + (size_t)(2 * vrow + 1) * HWPIX + jv * 4;
        const float4 a = *(const float4*)p0;  // k even -> low halves
        const float4 b = *(const float4*)p1;  // k odd  -> high halves
        int* dst = &xs32[(jv * 4) * XS_CSTRIDE + vrow];
        dst[0 * XS_CSTRIDE] = __builtin_bit_cast(int, __builtin_amdgcn_cvt_pkrtz(a.x, b.x));
        dst[1 * XS_CSTRIDE] = __builtin_bit_cast(int, __builtin_amdgcn_cvt_pkrtz(a.y, b.y));
        dst[2 * XS_CSTRIDE] = __builtin_bit_cast(int, __builtin_amdgcn_cvt_pkrtz(a.z, b.z));
        dst[3 * XS_CSTRIDE] = __builtin_bit_cast(int, __builtin_amdgcn_cvt_pkrtz(a.w, b.w));
    }
    __syncthreads();

    // ---- per-wave GEMM ----
    const int wave  = tid >> 5;               // 0..7 -> channel quadrant *64
    const int lane  = tid & 31;
    const int lhalf = lane & 15;
    const int lsel  = lane >> 4;              // 0: lanes 0-15, 1: lanes 16-31
    const int c0    = wave * 64;

    v8f acc[4][2] = {};                       // 64 VGPRs of f32 accumulation

    for (int kb = 0; kb < CCH; kb += 32) {
        const int vb = (kb >> 1) + lsel * 8;  // first k-pair word this lane needs

        // B fragments: two b128 loads each, words arrive in fragment order
        v16h bfrag[2];
        #pragma unroll
        for (int ph = 0; ph < 2; ++ph) {
            const int* src = &xs32[(ph * 16 + lhalf) * XS_CSTRIDE + vb];
            W8 t;
            t.a = *(const int4*)(src);        // K-pairs vb .. vb+3
            t.b = *(const int4*)(src + 4);    // K-pairs vb+4 .. vb+7
            bfrag[ph] = __builtin_bit_cast(v16h, t);
        }

        // A fragments, then 8 matrix ops
        v16h afrag[4];
        #pragma unroll
        for (int ct = 0; ct < 4; ++ct) {
            const int row = c0 + ct * 16 + lhalf;
            const _Float16* mr = M + (size_t)row * CCH + kb + lsel * 16;
            const v8h lo = *(const v8h*)(mr);        // K kb+0..7  / kb+16..23
            const v8h hi = *(const v8h*)(mr + 8);    // K kb+8..15 / kb+24..31
            #pragma unroll
            for (int i = 0; i < 8; ++i) { afrag[ct][i] = lo[i]; afrag[ct][8 + i] = hi[i]; }
        }

        #pragma unroll
        for (int ct = 0; ct < 4; ++ct) {
            #pragma unroll
            for (int ph = 0; ph < 2; ++ph) {
                acc[ct][ph] = __builtin_amdgcn_wmma_f32_16x16x32_f16(
                    /*neg_a=*/false, afrag[ct], /*neg_b=*/false, bfrag[ph],
                    /*c_mod=*/(short)0, acc[ct][ph],
                    /*reuse_a=*/false, /*reuse_b=*/false);
            }
        }
    }

    // ---- epilogue: C/D layout => VGPR r, lanes 0-15: M=r, lanes 16-31: M=r+8
    #pragma unroll
    for (int ct = 0; ct < 4; ++ct) {
        #pragma unroll
        for (int ph = 0; ph < 2; ++ph) {
            #pragma unroll
            for (int r = 0; r < 8; ++r) {
                const int ch = c0 + ct * 16 + r + lsel * 8;
                const int j  = ph * 16 + lhalf;
                Y[base + (size_t)ch * HWPIX + j] = acc[ct][ph][r];
            }
        }
    }
}

// ---------------------------------------------------------------------------
extern "C" void kernel_launch(void* const* d_in, const int* in_sizes, int n_in,
                              void* d_out, int out_size, void* d_ws, size_t ws_size,
                              hipStream_t stream) {
    const float* X   = (const float*)d_in[0];   // activations [32,512,56,56] f32
    const float* kin = (const float*)d_in[1];   // inhib_kernel [512] f32
    float* Y = (float*)d_out;

    float*    g = (float*)d_ws;                              // 512 f32
    _Float16* M = (_Float16*)((char*)d_ws + 4096);           // 512x512 f16

    build_inv_filter<<<1, 512, 0, stream>>>(kin, g);
    build_circulant<<<CCH, 256, 0, stream>>>(g, M);

    const int pixel_blocks = (32 * HWPIX) / 32;              // 3136
    circulant_gemm<<<pixel_blocks, 256, 0, stream>>>(X, M, Y);
}
